// YOLOv8n_MoE_C2f_19576460935692
// MI455X (gfx1250) — matrix operand order repaired and checked by reading, not compiled, not run on hardware
//
#include <hip/hip_runtime.h>
#include <hip/hip_bf16.h>

typedef __attribute__((ext_vector_type(16))) __bf16 v16bf;
typedef __attribute__((ext_vector_type(8)))  __bf16 v8bf;
typedef __attribute__((ext_vector_type(8)))  float  v8f;
typedef __attribute__((ext_vector_type(4)))  float  v4f;

#define CH   128
#define C1   256
#define C2   256
#define NTOK 32768
#define HW   4096

__device__ __forceinline__ float silu_f(float v) {
    return v / (1.0f + __expf(-v));
}

__device__ __forceinline__ v16bf cat16(v8bf lo, v8bf hi) {
    return __builtin_shufflevector(lo, hi, 0,1,2,3,4,5,6,7,8,9,10,11,12,13,14,15);
}

// ---------------------------------------------------------------------------
// Prep: convert weights to bf16 (W1, We center tap, W3)
// ---------------------------------------------------------------------------
__global__ __launch_bounds__(256) void k_prep(const float* __restrict__ W1,
                                              const float* __restrict__ We,
                                              const float* __restrict__ W3,
                                              __bf16* __restrict__ W1bf,
                                              __bf16* __restrict__ Wcbf,
                                              __bf16* __restrict__ W3bf) {
    int i = blockIdx.x * 256 + threadIdx.x;
    if (i < CH * C1) W1bf[i] = (__bf16)W1[i];
    int j = i - CH * C1;
    if (j >= 0 && j < 4 * CH * CH) Wcbf[j] = (__bf16)We[(size_t)j * 9 + 4];  // [e][o][c][1][1]
    int k = i - (CH * C1 + 4 * CH * CH);
    if (k >= 0 && k < C2 * CH) W3bf[k] = (__bf16)W3[k];
}

// ---------------------------------------------------------------------------
// Stage 1: xr[n][o] = silu( sum_c W1[o][c] * x[b][c][p] + b1[o] ), bf16 out
// Block: 16 tokens x 128 channels; 8 waves, one 16x16 tile each; K=256.
// ---------------------------------------------------------------------------
__global__ __launch_bounds__(256) void k_gemm1(const float*  __restrict__ x,
                                               const __bf16* __restrict__ W1bf,
                                               const float*  __restrict__ b1,
                                               __bf16*       __restrict__ xr) {
    __shared__ __attribute__((aligned(32))) __bf16 lx[16][C1];   // [token][k] 8 KB

    const int tokBase = blockIdx.x * 16;
    const int bidx = tokBase >> 12;      // / HW
    const int p0   = tokBase & (HW - 1);
    const float* xb = x + (size_t)bidx * C1 * HW + p0;

    // Stage x tile into LDS as bf16 [token][channel]
    {
        const int c = threadIdx.x;                      // 256 threads = 256 channels
        const float* src = xb + (size_t)c * HW;
        #pragma unroll
        for (int t = 0; t < 16; t += 4) {
            v4f v = *(const v4f*)(src + t);
            lx[t + 0][c] = (__bf16)v.x;
            lx[t + 1][c] = (__bf16)v.y;
            lx[t + 2][c] = (__bf16)v.z;
            lx[t + 3][c] = (__bf16)v.w;
        }
    }
    __syncthreads();

    const int wave   = threadIdx.x >> 5;
    const int lane   = threadIdx.x & 31;
    const int laneLo = lane & 15;
    const int laneHi = lane >> 4;
    const int oBase  = wave * 16;
    const int row    = oBase + laneLo;

    v8f acc = {0.f,0.f,0.f,0.f,0.f,0.f,0.f,0.f};
    #pragma unroll
    for (int k0 = 0; k0 < C1; k0 += 32) {
        const __bf16* ap = W1bf + (size_t)row * C1 + k0 + laneHi * 8;
        v16bf a = cat16(*(const v8bf*)ap, *(const v8bf*)(ap + 16));
        v16bf b = cat16(*(const v8bf*)&lx[laneLo][k0 + laneHi * 8], *(const v8bf*)&lx[laneLo][k0 + laneHi * 8 + 16]);
        acc = __builtin_amdgcn_wmma_f32_16x16x32_bf16(false, a, false, b,
                                                      (short)0, acc, false, false);
    }

    const int token = tokBase + laneLo;
    const int oB    = oBase + laneHi * 8;
    v8bf outv;
    #pragma unroll
    for (int r = 0; r < 8; ++r) {
        float v = acc[r] + b1[oB + r];
        outv[r] = (__bf16)silu_f(v);
    }
    *(v8bf*)(xr + (size_t)token * CH + oB) = outv;
}

// ---------------------------------------------------------------------------
// Stage 2: router — logits, softmax, top-2, renormalize -> dense wmat[n][4]
// ---------------------------------------------------------------------------
__global__ __launch_bounds__(256) void k_router(const __bf16* __restrict__ xr,
                                                const float*  __restrict__ Wr,
                                                const float*  __restrict__ br,
                                                float*        __restrict__ wmat) {
    __shared__ float wr[4][CH];
    for (int i = threadIdx.x; i < 4 * CH; i += 256) wr[i >> 7][i & 127] = Wr[i];
    __syncthreads();

    const int n = blockIdx.x * 256 + threadIdx.x;
    const __bf16* xp = xr + (size_t)n * CH;
    float l0 = br[0], l1 = br[1], l2 = br[2], l3 = br[3];
    #pragma unroll 8
    for (int c = 0; c < CH; ++c) {
        float xv = (float)xp[c];
        l0 += xv * wr[0][c];
        l1 += xv * wr[1][c];
        l2 += xv * wr[2][c];
        l3 += xv * wr[3][c];
    }
    float m = fmaxf(fmaxf(l0, l1), fmaxf(l2, l3));
    float p[4] = {__expf(l0 - m), __expf(l1 - m), __expf(l2 - m), __expf(l3 - m)};
    // top-1
    int i0 = 0; float v0 = p[0];
    #pragma unroll
    for (int e = 1; e < 4; ++e) if (p[e] > v0) { v0 = p[e]; i0 = e; }
    // top-2 (excluding i0)
    int i1 = -1; float v1 = -1.0f;
    #pragma unroll
    for (int e = 0; e < 4; ++e) if (e != i0 && p[e] > v1) { v1 = p[e]; i1 = e; }
    const float inv = 1.0f / (v0 + v1);      // softmax norm cancels in renorm
    float w[4] = {0.f, 0.f, 0.f, 0.f};
    w[i0] = v0 * inv;
    w[i1] = v1 * inv;
    #pragma unroll
    for (int e = 0; e < 4; ++e) wmat[(size_t)n * 4 + e] = w[e];
}

// ---------------------------------------------------------------------------
// Stage 3+4 fused: per 16-token tile
//   moe[n][o] = sum_e wmat[n][e] * silu(xr[n] . Wc[e][o] + be[e][o])   (8 waves x 16 ch)
//   -> LDS bf16 -> y[b][oc][p] = silu(moe[n] . W3[oc] + b3[oc]) + x    (8 waves x 32 ch)
// ---------------------------------------------------------------------------
__global__ __launch_bounds__(256) void k_moe_out(const __bf16* __restrict__ xr,
                                                 const __bf16* __restrict__ Wcbf,
                                                 const float*  __restrict__ be,
                                                 const float*  __restrict__ wmat,
                                                 const __bf16* __restrict__ W3bf,
                                                 const float*  __restrict__ b3,
                                                 const float*  __restrict__ x,
                                                 float*        __restrict__ y) {
    __shared__ __attribute__((aligned(32))) __bf16 lmoe[16][CH];  // [token][ch] 4 KB

    const int tokBase = blockIdx.x * 16;
    const int wave   = threadIdx.x >> 5;
    const int lane   = threadIdx.x & 31;
    const int laneLo = lane & 15;
    const int laneHi = lane >> 4;
    const int token  = tokBase + laneLo;
    const int oBase  = wave * 16;
    const int row    = oBase + laneLo;

    float wl[4];
    #pragma unroll
    for (int e = 0; e < 4; ++e) wl[e] = wmat[(size_t)token * 4 + e];

    v8f moe = {0.f,0.f,0.f,0.f,0.f,0.f,0.f,0.f};
    #pragma unroll
    for (int e = 0; e < 4; ++e) {
        const __bf16* Wce = Wcbf + (size_t)e * CH * CH;
        v8f acc = {0.f,0.f,0.f,0.f,0.f,0.f,0.f,0.f};
        #pragma unroll
        for (int k0 = 0; k0 < CH; k0 += 32) {
            const __bf16* ap = Wce + (size_t)row * CH + k0 + laneHi * 8;
            v16bf a = cat16(*(const v8bf*)ap, *(const v8bf*)(ap + 16));
            v16bf b = cat16(*(const v8bf*)(xr + (size_t)token * CH + k0 + laneHi * 8), *(const v8bf*)(xr + (size_t)token * CH + k0 + laneHi * 8 + 16));
            acc = __builtin_amdgcn_wmma_f32_16x16x32_bf16(false, a, false, b,
                                                          (short)0, acc, false, false);
        }
        const int oB = oBase + laneHi * 8;
        #pragma unroll
        for (int r = 0; r < 8; ++r) {
            float h = acc[r] + be[e * CH + oB + r];
            moe[r] += wl[e] * silu_f(h);
        }
    }

    {   // moe tile -> LDS bf16 (token-major)
        const int oB = oBase + laneHi * 8;
        v8bf mv;
        #pragma unroll
        for (int r = 0; r < 8; ++r) mv[r] = (__bf16)moe[r];
        *(v8bf*)&lmoe[laneLo][oB] = mv;
    }
    __syncthreads();

    // Stage 4: each wave covers 32 output channels (two 16x16 tiles)
    const int bidx = tokBase >> 12;
    const int p0   = tokBase & (HW - 1);
    const float* xbase = x + (size_t)bidx * C2 * HW + p0 + laneLo;
    float*       ybase = y + (size_t)bidx * C2 * HW + p0 + laneLo;

    #pragma unroll
    for (int tIdx = 0; tIdx < 2; ++tIdx) {
        const int ocBase = wave * 32 + tIdx * 16;
        const int orow   = ocBase + laneLo;
        v8f acc = {0.f,0.f,0.f,0.f,0.f,0.f,0.f,0.f};
        #pragma unroll
        for (int k0 = 0; k0 < CH; k0 += 32) {
            const __bf16* ap = W3bf + (size_t)orow * CH + k0 + laneHi * 8;
            v16bf a = cat16(*(const v8bf*)ap, *(const v8bf*)(ap + 16));
            v16bf b = cat16(*(const v8bf*)&lmoe[laneLo][k0 + laneHi * 8], *(const v8bf*)&lmoe[laneLo][k0 + laneHi * 8 + 16]);
            acc = __builtin_amdgcn_wmma_f32_16x16x32_bf16(false, a, false, b,
                                                          (short)0, acc, false, false);
        }
        const int oB = ocBase + laneHi * 8;
        #pragma unroll
        for (int r = 0; r < 8; ++r) {
            const int o = oB + r;
            float v = silu_f(acc[r] + b3[o]) + xbase[(size_t)o * HW];
            ybase[(size_t)o * HW] = v;
        }
    }
}

// ---------------------------------------------------------------------------
extern "C" void kernel_launch(void* const* d_in, const int* in_sizes, int n_in,
                              void* d_out, int out_size, void* d_ws, size_t ws_size,
                              hipStream_t stream) {
    const float* x  = (const float*)d_in[0];   // (8,256,64,64)
    const float* W1 = (const float*)d_in[1];   // (128,256)
    const float* b1 = (const float*)d_in[2];   // (128,)
    const float* Wr = (const float*)d_in[3];   // (4,128)
    const float* br = (const float*)d_in[4];   // (4,)
    const float* We = (const float*)d_in[5];   // (4,128,128,3,3)
    const float* be = (const float*)d_in[6];   // (4,128)
    const float* W3 = (const float*)d_in[7];   // (256,128)
    const float* b3 = (const float*)d_in[8];   // (256,)
    float* y = (float*)d_out;

    char* ws = (char*)d_ws;
    __bf16* W1bf = (__bf16*)(ws);                       //  64 KB
    __bf16* Wcbf = (__bf16*)(ws + (64 << 10));          // 128 KB
    __bf16* W3bf = (__bf16*)(ws + (192 << 10));         //  64 KB
    __bf16* xr   = (__bf16*)(ws + (256 << 10));         //   8 MB
    float*  wmat = (float*)(ws + (256 << 10) + ((size_t)NTOK * CH * 2)); // 512 KB

    k_prep   <<<512, 256, 0, stream>>>(W1, We, W3, W1bf, Wcbf, W3bf);
    k_gemm1  <<<NTOK / 16, 256, 0, stream>>>(x, W1bf, b1, xr);
    k_router <<<NTOK / 256, 256, 0, stream>>>(xr, Wr, br, wmat);
    k_moe_out<<<NTOK / 16, 256, 0, stream>>>(xr, Wcbf, be, wmat, W3bf, b3, x, y);
}
